// SuzgunStackRNN_38328288150272
// MI455X (gfx1250) — compile-verified, matrix operand order checked by reading, not run on hardware
//
#include <hip/hip_runtime.h>
#include <hip/hip_bf16.h>

#define HID 256
#define DEP 128
#define VD  8
#define VOC 512
#define NB  64
#define TT  2048

typedef __bf16 bf16;
typedef __attribute__((ext_vector_type(16))) __bf16 v16bf;
typedef __attribute__((ext_vector_type(8)))  __bf16 v8bf;
typedef __attribute__((ext_vector_type(8)))  float  v8f;

__device__ __forceinline__ bf16 f2bf(float f) { return (bf16)f; }

__device__ __forceinline__ v16bf join8(v8bf lo, v8bf hi) {
  v16bf r;
#pragma unroll
  for (int i = 0; i < 8; ++i) { r[i] = lo[i]; r[i + 8] = hi[i]; }
  return r;
}

// ------------- Kernel 0: weight conversion + Wc = W_hh@W_sh + stack init -------------
__global__ void k_prep(const float* __restrict__ W_ih, const float* __restrict__ W_hh,
                       const float* __restrict__ W_y,  const float* __restrict__ stack0,
                       const float* __restrict__ W_sh,
                       bf16* __restrict__ Wih_bf, bf16* __restrict__ Whh_bf,
                       bf16* __restrict__ Wy_bf,  float* __restrict__ sbuf,
                       float* __restrict__ Wc) {
  const int n_ih = HID * VOC, n_hh = HID * HID, n_y = VOC * HID, n_s = NB * DEP * VD;
  int i = blockIdx.x * blockDim.x + threadIdx.x;
  if (i < n_ih)                   Wih_bf[i] = f2bf(W_ih[i]);
  else if (i < n_ih + n_hh)       Whh_bf[i - n_ih] = f2bf(W_hh[i - n_ih]);
  else if (i < n_ih + n_hh + n_y) Wy_bf[i - n_ih - n_hh] = f2bf(W_y[i - n_ih - n_hh]);
  else if (i < n_ih + n_hh + n_y + n_s) {
    int j = i - n_ih - n_hh - n_y;
    sbuf[j] = stack0[j];
  } else if (i < n_ih + n_hh + n_y + n_s + HID * VD) {
    int o = i - n_ih - n_hh - n_y - n_s;
    int n = o >> 3, v = o & 7;
    float s = 0.f;
    for (int k = 0; k < HID; ++k) s += W_hh[n * HID + k] * W_sh[k * VD + v];
    Wc[o] = s;
  }
}

// ------------- Kernel 1: Z = inputs @ W_ih^T  (M=B*T, N=256, K=512) -------------
__global__ void __launch_bounds__(256)
k_proj_in(const float* __restrict__ X, const bf16* __restrict__ Wb, float* __restrict__ Z) {
  const int m0   = blockIdx.x * 16;
  const int lane = threadIdx.x & 31;
  const int w    = threadIdx.x >> 5;
  const int mr    = m0 + (lane & 15);
  const int khalf = (lane >> 4) * 8;
  const int nb    = w * 32 + (lane & 15);
  v8f acc0 = {}, acc1 = {};
  for (int ks = 0; ks < VOC; ks += 32) {
    int kb = ks + khalf;
    const float* ap = X + (size_t)mr * VOC + kb;
    v16bf a;
#pragma unroll
    for (int i = 0; i < 8; ++i) { a[i] = f2bf(ap[i]); a[i + 8] = f2bf(ap[i + 16]); }
    v8bf b0lo = *(const v8bf*)(Wb + (size_t)nb * VOC + kb);
    v8bf b0hi = *(const v8bf*)(Wb + (size_t)nb * VOC + kb + 16);
    v8bf b1lo = *(const v8bf*)(Wb + (size_t)(nb + 16) * VOC + kb);
    v8bf b1hi = *(const v8bf*)(Wb + (size_t)(nb + 16) * VOC + kb + 16);
    acc0 = __builtin_amdgcn_wmma_f32_16x16x32_bf16(false, a, false, join8(b0lo, b0hi),
                                                   (short)0, acc0, false, false);
    acc1 = __builtin_amdgcn_wmma_f32_16x16x32_bf16(false, a, false, join8(b1lo, b1hi),
                                                   (short)0, acc1, false, false);
  }
#pragma unroll
  for (int r = 0; r < 8; ++r) {
    int m = m0 + r + khalf;
    Z[(size_t)m * HID + nb]      = acc0[r];
    Z[(size_t)m * HID + nb + 16] = acc1[r];
  }
}

// ------------- Kernel 2: recurrent scan, 4 blocks x 16 batch rows -------------
__global__ void __launch_bounds__(512)
k_rnn(const float* __restrict__ Z, const bf16* __restrict__ Whh,
      const float* __restrict__ b_ih, const float* __restrict__ b_hh,
      const float* __restrict__ Wc,  const float* __restrict__ W_a,
      const float* __restrict__ W_n, const float* __restrict__ hidden0,
      float* __restrict__ sbuf, bf16* __restrict__ Hbf, float* __restrict__ out_tail) {
  __shared__ float h[16 * HID];      // hidden state (fp32)
  __shared__ float wa[3 * HID];
  __shared__ float wn[VD * HID];
  __shared__ float stkt[16 * VD];
  __shared__ float alog[16 * 3];
  __shared__ float nsg[16 * VD];
  __shared__ float pdot[256 + 96];

  const int tid   = threadIdx.x;
  const int bbase = blockIdx.x * 16;

  for (int i = tid; i < 16 * HID; i += 512) h[i] = hidden0[bbase * HID + i];
  for (int i = tid; i < 3 * HID;  i += 512) wa[i] = W_a[i];
  for (int i = tid; i < VD * HID; i += 512) wn[i] = W_n[i];

  const int lane  = tid & 31;
  const int w     = tid >> 5;          // 16 waves, one 16x16 N-tile each
  const int khalf = (lane >> 4) * 8;
  const int mloc  = lane & 15;
  const int n0    = w * 16 + (lane & 15);
  const float bn  = b_ih[n0] + b_hh[n0];

  // time-invariant W_hh B-fragments -> registers (64 VGPRs)
  v16bf bfr[8];
#pragma unroll
  for (int ks8 = 0; ks8 < 8; ++ks8) {
    int kb = ks8 * 32 + khalf;
    v8bf blo = *(const v8bf*)(Whh + (size_t)n0 * HID + kb);
    v8bf bhi = *(const v8bf*)(Whh + (size_t)n0 * HID + kb + 16);
    bfr[ks8] = join8(blo, bhi);
  }
  // stack-top correction weights Wc = W_hh@W_sh, per-lane column n0
  float wc[VD];
#pragma unroll
  for (int v = 0; v < VD; ++v) wc[v] = Wc[n0 * VD + v];

  const size_t SB = (size_t)NB * DEP * VD;
  __syncthreads();

  for (int t = 0; t < TT; ++t) {
    const float* cur = sbuf + (size_t)(t & 1) * SB;
    float*       nxt = sbuf + (size_t)((t + 1) & 1) * SB;
    // stack top -> LDS; Z -> regs (overlaps the WMMA chain); prefetch next Z slice
    if (tid < 16 * VD)
      stkt[tid] = cur[(size_t)(bbase + (tid >> 3)) * DEP * VD + (tid & 7)];
    float zv[8];
#pragma unroll
    for (int r = 0; r < 8; ++r)
      zv[r] = Z[((size_t)(bbase + r + khalf) * TT + t) * HID + n0];
    if (t + 1 < TT) {
#pragma unroll
      for (int r = 0; r < 8; ++r)
        __builtin_prefetch(&Z[((size_t)(bbase + r + khalf) * TT + t + 1) * HID + n0], 0, 1);
    }
    __syncthreads();
    // pre = hidden @ W_hh^T : two independent accumulation chains
    v8f acc0 = {}, acc1 = {};
#pragma unroll
    for (int ks8 = 0; ks8 < 8; ks8 += 2) {
      int kb = ks8 * 32 + khalf;
      v16bf a0, a1;
#pragma unroll
      for (int i = 0; i < 8; ++i) {
        a0[i]     = f2bf(h[mloc * HID + kb + i]);
        a0[i + 8] = f2bf(h[mloc * HID + kb + 16 + i]);
        a1[i]     = f2bf(h[mloc * HID + kb + 32 + i]);
        a1[i + 8] = f2bf(h[mloc * HID + kb + 48 + i]);
      }
      acc0 = __builtin_amdgcn_wmma_f32_16x16x32_bf16(false, a0, false, bfr[ks8],
                                                     (short)0, acc0, false, false);
      acc1 = __builtin_amdgcn_wmma_f32_16x16x32_bf16(false, a1, false, bfr[ks8 + 1],
                                                     (short)0, acc1, false, false);
    }
    __syncthreads();   // all reads of hidden complete
    // hidden_new = tanh(pre + stack_corr + Z + bias)
#pragma unroll
    for (int r = 0; r < 8; ++r) {
      int ml = r + khalf;
      float corr = 0.f;
#pragma unroll
      for (int v = 0; v < VD; ++v) corr += stkt[ml * VD + v] * wc[v];
      float val = tanhf(acc0[r] + acc1[r] + corr + zv[r] + bn);
      h[ml * HID + n0] = val;
      Hbf[((size_t)(bbase + ml) * TT + t) * HID + n0] = f2bf(val);
    }
    __syncthreads();
    // head partial dots (half-length, bank-rotated)
    if (tid < 256) {                       // n: half*128 + v*16 + b
      int b = tid & 15, v = (tid >> 4) & 7, half = tid >> 7;
      int base = half * 128;
      float s = 0.f;
#pragma unroll 4
      for (int i = 0; i < 128; ++i) {
        int hh = base + ((i + b * 8) & 127);
        s += h[b * HID + hh] * wn[v * HID + hh];
      }
      pdot[tid] = s;
    } else if (tid < 352) {                // a: half*48 + j*16 + b
      int x = tid - 256;
      int b = x & 15, r2 = x >> 4;
      int j = r2 % 3, half = r2 / 3;
      int base = half * 128;
      float s = 0.f;
#pragma unroll 4
      for (int i = 0; i < 128; ++i) {
        int hh = base + ((i + b * 8) & 127);
        s += h[b * HID + hh] * wa[j * HID + hh];
      }
      pdot[256 + x] = s;
    }
    __syncthreads();
    // stack neighbor rows load (independent of heads -> overlaps this phase)
    const int sb = tid >> 5;               // local batch row
    const int d0 = (tid & 31) * 4;         // 4 depth rows per thread
    size_t cbase = ((size_t)(bbase + sb) * DEP + d0) * VD;
    float rows[6][VD];
#pragma unroll
    for (int d = 0; d < 4; ++d)
#pragma unroll
      for (int v = 0; v < VD; ++v) rows[d + 1][v] = cur[cbase + d * VD + v];
#pragma unroll
    for (int v = 0; v < VD; ++v) {
      rows[0][v] = (d0 == 0)       ? 0.f : cur[cbase - VD + v];
      rows[5][v] = (d0 + 4 >= DEP) ? 0.f : cur[cbase + 4 * VD + v];
    }
    // combine partials: sigmoid for n, fused softmax for a
    if (tid < 128) {
      int v2 = tid >> 4, b2 = tid & 15;
      float s = pdot[tid] + pdot[tid + 128];
      nsg[b2 * VD + v2] = 1.f / (1.f + __expf(-s));
    } else if (tid < 144) {
      int b2 = tid - 128;
      float l0 = pdot[256 + b2]      + pdot[256 + 48 + b2];
      float l1 = pdot[256 + 16 + b2] + pdot[256 + 64 + b2];
      float l2 = pdot[256 + 32 + b2] + pdot[256 + 80 + b2];
      float m = fmaxf(l0, fmaxf(l1, l2));
      float e0 = __expf(l0 - m), e1 = __expf(l1 - m), e2 = __expf(l2 - m);
      float inv = 1.f / (e0 + e1 + e2);
      alog[b2 * 3] = e0 * inv; alog[b2 * 3 + 1] = e1 * inv; alog[b2 * 3 + 2] = e2 * inv;
    }
    __syncthreads();
    // stack blend -> other parity buffer
    {
      float a0 = alog[sb * 3], a1 = alog[sb * 3 + 1], a2 = alog[sb * 3 + 2];
      if (d0 == 0) {
#pragma unroll
        for (int v = 0; v < VD; ++v) rows[0][v] = nsg[sb * VD + v];
      }
#pragma unroll
      for (int d = 0; d < 4; ++d)
#pragma unroll
        for (int v = 0; v < VD; ++v)
          nxt[cbase + d * VD + v] = a0 * rows[d][v] + a1 * rows[d + 2][v] + a2 * rows[d + 1][v];
    }
    __threadfence();
    __syncthreads();
  }

  // final stack (parity TT&1 == 0) and hidden into d_out tail
  for (int i = tid; i < 16 * DEP * VD; i += 512)
    out_tail[bbase * DEP * VD + i] = sbuf[(size_t)bbase * DEP * VD + i];
  for (int i = tid; i < 16 * HID; i += 512)
    out_tail[NB * DEP * VD + bbase * HID + i] = h[i];
}

// ------------- Kernel 3: outputs = hidden_all @ W_y^T  (M=B*T, N=512, K=256) -------------
__global__ void __launch_bounds__(256)
k_proj_out(const bf16* __restrict__ Hb, const bf16* __restrict__ Wy, float* __restrict__ Y) {
  const int m0   = blockIdx.x * 16;
  const int lane = threadIdx.x & 31;
  const int w    = threadIdx.x >> 5;
  const int mr    = m0 + (lane & 15);
  const int khalf = (lane >> 4) * 8;
  const int nb    = w * 64 + (lane & 15);
  v8f acc0 = {}, acc1 = {}, acc2 = {}, acc3 = {};
  for (int ks = 0; ks < HID; ks += 32) {
    int kb = ks + khalf;
    v8bf alo = *(const v8bf*)(Hb + (size_t)mr * HID + kb);
    v8bf ahi = *(const v8bf*)(Hb + (size_t)mr * HID + kb + 16);
    v16bf a = join8(alo, ahi);
#pragma unroll
    for (int q = 0; q < 4; ++q) {
      int n = nb + q * 16;
      v8bf blo = *(const v8bf*)(Wy + (size_t)n * HID + kb);
      v8bf bhi = *(const v8bf*)(Wy + (size_t)n * HID + kb + 16);
      v16bf b = join8(blo, bhi);
      v8f* acc = (q == 0) ? &acc0 : (q == 1) ? &acc1 : (q == 2) ? &acc2 : &acc3;
      *acc = __builtin_amdgcn_wmma_f32_16x16x32_bf16(false, a, false, b,
                                                     (short)0, *acc, false, false);
    }
  }
#pragma unroll
  for (int r = 0; r < 8; ++r) {
    int m = m0 + r + khalf;
    Y[(size_t)m * VOC + nb]      = acc0[r];
    Y[(size_t)m * VOC + nb + 16] = acc1[r];
    Y[(size_t)m * VOC + nb + 32] = acc2[r];
    Y[(size_t)m * VOC + nb + 48] = acc3[r];
  }
}

extern "C" void kernel_launch(void* const* d_in, const int* in_sizes, int n_in,
                              void* d_out, int out_size, void* d_ws, size_t ws_size,
                              hipStream_t stream) {
  const float* inputs  = (const float*)d_in[0];
  const float* stack0  = (const float*)d_in[1];
  const float* hidden0 = (const float*)d_in[2];
  const float* W_ih    = (const float*)d_in[3];
  const float* b_ih    = (const float*)d_in[4];
  const float* W_hh    = (const float*)d_in[5];
  const float* b_hh    = (const float*)d_in[6];
  const float* W_sh    = (const float*)d_in[7];
  const float* W_y     = (const float*)d_in[8];
  const float* W_a     = (const float*)d_in[9];
  const float* W_n     = (const float*)d_in[10];

  char* ws = (char*)d_ws;
  size_t off = 0;
  bf16*  Wih_bf = (bf16*)(ws + off);  off += (size_t)HID * VOC * sizeof(bf16);
  bf16*  Whh_bf = (bf16*)(ws + off);  off += (size_t)HID * HID * sizeof(bf16);
  bf16*  Wy_bf  = (bf16*)(ws + off);  off += (size_t)VOC * HID * sizeof(bf16);
  float* Wc     = (float*)(ws + off); off += (size_t)HID * VD * sizeof(float);
  float* sbuf   = (float*)(ws + off); off += (size_t)2 * NB * DEP * VD * sizeof(float);
  float* Z      = (float*)(ws + off); off += (size_t)NB * TT * HID * sizeof(float);
  bf16*  Hbf    = (bf16*)(ws + off);  off += (size_t)NB * TT * HID * sizeof(bf16);

  float* Y        = (float*)d_out;
  float* out_tail = Y + (size_t)NB * TT * VOC;

  const int prep_n = HID * VOC + HID * HID + VOC * HID + NB * DEP * VD + HID * VD;
  k_prep<<<(prep_n + 255) / 256, 256, 0, stream>>>(W_ih, W_hh, W_y, stack0, W_sh,
                                                   Wih_bf, Whh_bf, Wy_bf, sbuf, Wc);
  k_proj_in<<<NB * TT / 16, 256, 0, stream>>>(inputs, Wih_bf, Z);
  k_rnn<<<NB / 16, 512, 0, stream>>>(Z, Whh_bf, b_ih, b_hh, Wc, W_a, W_n,
                                     hidden0, sbuf, Hbf, out_tail);
  k_proj_out<<<NB * TT / 16, 256, 0, stream>>>(Hbf, Wy_bf, Y);
}
